// SimpleAttention_58789512348123
// MI455X (gfx1250) — matrix operand order repaired and checked by reading, not compiled, not run on hardware
//
#include <hip/hip_runtime.h>

typedef __attribute__((ext_vector_type(4)))  _Float16 h4;
typedef __attribute__((ext_vector_type(8)))  _Float16 v8h;
typedef __attribute__((ext_vector_type(16))) _Float16 v16h;
typedef __attribute__((ext_vector_type(8)))  float    v8f;
typedef __attribute__((ext_vector_type(4)))  float    v4f;

constexpr int kChunk = 64;
constexpr int kC     = 128;
constexpr int kQKV   = 384;

// Padded LDS strides (in f16 elements): +8 f16 (=16B) rotates banks by 4 per
// row -> wave-wide b128 row accesses achieve the 2-pass LDS optimum.
constexpr int kSxs = 136;  // for 128-wide matrices
constexpr int kSvs = 72;   // for 64-wide matrices

// LDS element offsets (in _Float16 units)
constexpr int kOffWt   = 0;                      // Wqkv^T : 384 x kSxs
constexpr int kOffWpt  = kOffWt + kQKV * kSxs;   // Wproj^T: 128 x kSxs
constexpr int kOffGrp  = kOffWpt + kC * kSxs;    // two per-chunk groups follow
constexpr int kGXO = 0;                          // x (phase0/1) aliased with O: 64 x kSxs
constexpr int kGQ  = kGXO + kChunk * kSxs;       // Q : 64 x kSxs (row-major)
constexpr int kGK  = kGQ  + kChunk * kSxs;       // K : 64 x kSxs (row-major)
constexpr int kGVt = kGK  + kChunk * kSxs;       // V^T : 128 x kSvs
constexpr int kGP  = kGVt + kC * kSvs;           // P : 64 x kSvs (row-major)
constexpr int kGrpSz = kGP + kChunk * kSvs;
constexpr int kSmemElems = kOffGrp + 2 * kGrpSz;
constexpr size_t kSmemBytes = size_t(kSmemElems) * sizeof(_Float16); // 299008 B

// A fragment: lane l<16 holds row M=l, K = {kb+0..7, kb+16..23}; lanes 16..31
// hold the same row with K = {kb+8..15, kb+24..31}  (CDNA5 16-bit A layout).
__device__ __forceinline__ v16h frag_a(const _Float16* row, int kb, int hi) {
  const _Float16* p = row + kb + (hi << 3);
  v8h lo = *(const v8h*)(p);
  v8h hh = *(const v8h*)(p + 16);
  v16h r;
#pragma unroll
  for (int i = 0; i < 8; ++i) { r[i] = lo[i]; r[i + 8] = hh[i]; }
  return r;
}

// B fragment (from B^T stored row-major): lane l<16 holds column N=l with
// K = {kb+0..7, kb+16..23}; lanes 16..31 hold K = {kb+8..15, kb+24..31}.
__device__ __forceinline__ v16h frag_b(const _Float16* row, int kb, int hi) {
  const _Float16* p = row + kb + (hi << 3);
  v8h lo = *(const v8h*)(p);
  v8h hh = *(const v8h*)(p + 16);
  v16h r;
#pragma unroll
  for (int i = 0; i < 8; ++i) { r[i] = lo[i]; r[i + 8] = hh[i]; }
  return r;
}

__device__ __forceinline__ v8f wmma_f16(v16h a, v16h b, v8f c) {
  return __builtin_amdgcn_wmma_f32_16x16x32_f16(false, a, false, b, (short)0, c,
                                                false, false);
}

__global__ __launch_bounds__(256) void block_attn_kernel(
    const float* __restrict__ x, const float* __restrict__ Wqkv,
    const float* __restrict__ bqkv, const float* __restrict__ Wp,
    const float* __restrict__ bp, float* __restrict__ out) {
  extern __shared__ __align__(16) char smem_raw[];
  _Float16* smem = (_Float16*)smem_raw;
  _Float16* sWt  = smem + kOffWt;
  _Float16* sWpt = smem + kOffWpt;

  const int tid  = threadIdx.x;
  const int wv   = tid >> 5;
  const int lane = tid & 31;
  const int g    = wv >> 2;        // which chunk of the pair
  const int mw   = wv & 3;         // 16-row M tile within chunk
  const int l15  = lane & 15;
  const int hi   = lane >> 4;
  const int rm   = mw << 4;

  _Float16* grp = smem + kOffGrp + g * kGrpSz;
  _Float16* sX  = grp + kGXO;      // aliased with sO
  _Float16* sQ  = grp + kGQ;
  _Float16* sK  = grp + kGK;
  _Float16* sVt = grp + kGVt;
  _Float16* sP  = grp + kGP;
  _Float16* sO  = grp + kGXO;

  // ---------- Phase 0: cooperative loads + f32->f16 conversion ----------
  for (int qi = tid; qi < (kC / 4) * kQKV; qi += 256) {
    const int n  = qi % kQKV;
    const int k0 = (qi / kQKV) << 2;
    h4 pk;
#pragma unroll
    for (int j = 0; j < 4; ++j) pk[j] = (_Float16)Wqkv[(k0 + j) * kQKV + n];
    *(h4*)(sWt + n * kSxs + k0) = pk;
  }
  for (int qi = tid; qi < (kC / 4) * kC; qi += 256) {
    const int n  = qi & (kC - 1);
    const int k0 = (qi / kC) << 2;
    h4 pk;
#pragma unroll
    for (int j = 0; j < 4; ++j) pk[j] = (_Float16)Wp[(k0 + j) * kC + n];
    *(h4*)(sWpt + n * kSxs + k0) = pk;
  }
  {
    const float* xb = x + (size_t)blockIdx.x * 2 * kChunk * kC;
    _Float16* sX0 = smem + kOffGrp + kGXO;
    _Float16* sX1 = smem + kOffGrp + kGrpSz + kGXO;
    for (int i = tid * 4; i < 2 * kChunk * kC; i += 256 * 4) {
      v4f w = *(const v4f*)(xb + i);
      const int ci = i & (kChunk * kC - 1);
      const int r = ci >> 7, c = ci & 127;
      h4 pk;
#pragma unroll
      for (int j = 0; j < 4; ++j) pk[j] = (_Float16)w[j];
      *(h4*)(((i < kChunk * kC) ? sX0 : sX1) + r * kSxs + c) = pk;
    }
  }
  __syncthreads();

  // ---------- Phase 1a: Q,K via (qkv)^T = Wqkv^T @ x^T ----------
  // Manual 2-tile software pipeline: acc0 and acc1 WMMA chains issue
  // back-to-back, then both epilogues run -> tile1's WMMAs fill tile0's
  // WMMA->VALU hazard window and vice versa (no v_nops).
  {
    v16h bx[4];
#pragma unroll
    for (int kq = 0; kq < 4; ++kq)
      bx[kq] = frag_b(sX + (rm + l15) * kSxs, kq * 32, hi);

    auto qk_tile = [&](int dt) {
      const int d0 = dt << 4;
      v8f acc = {};
#pragma unroll
      for (int kq = 0; kq < 4; ++kq) {
        v16h a = frag_a(sWt + (d0 + l15) * kSxs, kq * 32, hi);
        acc = wmma_f16(a, bx[kq], acc);
      }
      return acc;
    };
    auto qk_epi = [&](int dt, v8f acc) {
      const int d0 = dt << 4;
      v4f b0 = *(const v4f*)(bqkv + d0 + (hi << 3));
      v4f b1 = *(const v4f*)(bqkv + d0 + (hi << 3) + 4);
      v8h pk;
#pragma unroll
      for (int r = 0; r < 4; ++r) pk[r] = (_Float16)(acc[r] + b0[r]);
#pragma unroll
      for (int r = 0; r < 4; ++r) pk[r + 4] = (_Float16)(acc[r + 4] + b1[r]);
      _Float16* dst = (dt < 8 ? sQ : sK) + (rm + l15) * kSxs +
                      ((dt & 7) << 4) + (hi << 3);
      *(v8h*)dst = pk;
    };

    for (int dt = 0; dt < 16; dt += 2) {       // d-tiles: Q (0..7), K (8..15)
      v8f a0 = qk_tile(dt);
      v8f a1 = qk_tile(dt + 1);
      qk_epi(dt, a0);
      qk_epi(dt + 1, a1);
    }
  }
  // ---------- Phase 1b: V in original orientation -> V^T ----------
  {
    v16h ax[4];
#pragma unroll
    for (int kq = 0; kq < 4; ++kq)
      ax[kq] = frag_a(sX + (rm + l15) * kSxs, kq * 32, hi);

    auto v_tile = [&](int ct) {
      const int c0 = ct << 4;
      v8f acc = {};
#pragma unroll
      for (int kq = 0; kq < 4; ++kq) {
        v16h b = frag_b(sWt + (256 + c0 + l15) * kSxs, kq * 32, hi);
        acc = wmma_f16(ax[kq], b, acc);
      }
      return acc;
    };
    auto v_epi = [&](int ct, v8f acc) {
      const int c0 = ct << 4;
      const float bias = bqkv[256 + c0 + l15];
      v8h pk;
#pragma unroll
      for (int r = 0; r < 8; ++r) pk[r] = (_Float16)(acc[r] + bias);
      *(v8h*)(sVt + (c0 + l15) * kSvs + rm + (hi << 3)) = pk;
    };

    for (int ct = 0; ct < 8; ct += 2) {
      v8f a0 = v_tile(ct);
      v8f a1 = v_tile(ct + 1);
      v_epi(ct, a0);
      v_epi(ct + 1, a1);
    }
  }
  __syncthreads();

  // ---------- Phase 2: S^T = K @ "Q^T", softmax over t (M dim) ----------
  {
    v16h bq[4];
#pragma unroll
    for (int kq = 0; kq < 4; ++kq)
      bq[kq] = frag_b(sQ + (rm + l15) * kSxs, kq * 32, hi);

    v8f S[4];
#pragma unroll
    for (int tt = 0; tt < 4; ++tt) {
      v8f acc = {};
#pragma unroll
      for (int kq = 0; kq < 4; ++kq) {
        v16h a = frag_a(sK + (tt * 16 + l15) * kSxs, kq * 32, hi);
        acc = wmma_f16(a, bq[kq], acc);
      }
      S[tt] = acc;
    }

    const float scale = 0.0883883476483184f; // 128^-0.5
    float mx = S[0][0];
#pragma unroll
    for (int tt = 0; tt < 4; ++tt)
#pragma unroll
      for (int r = 0; r < 8; ++r) mx = fmaxf(mx, S[tt][r]);
    mx = fmaxf(mx, __shfl_xor(mx, 16));

    float sum = 0.f;
#pragma unroll
    for (int tt = 0; tt < 4; ++tt)
#pragma unroll
      for (int r = 0; r < 8; ++r) {
        const float e = __expf((S[tt][r] - mx) * scale);
        S[tt][r] = e;
        sum += e;
      }
    sum += __shfl_xor(sum, 16);
    const float inv = 1.0f / sum;

#pragma unroll
    for (int tt = 0; tt < 4; ++tt) {
      v8h pk;
#pragma unroll
      for (int r = 0; r < 8; ++r) pk[r] = (_Float16)(S[tt][r] * inv);
      *(v8h*)(sP + (rm + l15) * kSvs + tt * 16 + (hi << 3)) = pk;
    }
  }
  __syncthreads();

  // ---------- Phase 3: O^T = V^T @ "P^T" -> row-major O (b128 stores) ----
  {
    v16h bpv[2];
#pragma unroll
    for (int kq = 0; kq < 2; ++kq)
      bpv[kq] = frag_b(sP + (rm + l15) * kSvs, kq * 32, hi);

    auto o_tile = [&](int ct) {
      v8f acc = {};
#pragma unroll
      for (int kq = 0; kq < 2; ++kq) {
        v16h a = frag_a(sVt + (ct * 16 + l15) * kSvs, kq * 32, hi);
        acc = wmma_f16(a, bpv[kq], acc);
      }
      return acc;
    };
    auto o_epi = [&](int ct, v8f acc) {
      v8h pk;
#pragma unroll
      for (int r = 0; r < 8; ++r) pk[r] = (_Float16)acc[r];
      *(v8h*)(sO + (rm + l15) * kSxs + ct * 16 + (hi << 3)) = pk;
    };

    for (int ct = 0; ct < 8; ct += 2) {
      v8f a0 = o_tile(ct);
      v8f a1 = o_tile(ct + 1);
      o_epi(ct, a0);
      o_epi(ct + 1, a1);
    }
  }
  __syncthreads();

  // ---------- Phase 4: Y^T = Wproj^T @ "O^T" + b -> global b128 stores ----
  {
    v16h bo[4];
#pragma unroll
    for (int kq = 0; kq < 4; ++kq)
      bo[kq] = frag_b(sO + (rm + l15) * kSxs, kq * 32, hi);

    const size_t chunk = (size_t)blockIdx.x * 2 + g;
    float* outp = out + chunk * (kChunk * kC);

    auto y_tile = [&](int nt) {
      const int n0 = nt << 4;
      v8f acc = {};
#pragma unroll
      for (int kq = 0; kq < 4; ++kq) {
        v16h a = frag_a(sWpt + (n0 + l15) * kSxs, kq * 32, hi);
        acc = wmma_f16(a, bo[kq], acc);
      }
      return acc;
    };
    auto y_epi = [&](int nt, v8f acc) {
      const int n0 = nt << 4;
      v4f c0 = *(const v4f*)(bp + n0 + (hi << 3));
      v4f c1 = *(const v4f*)(bp + n0 + (hi << 3) + 4);
      v4f o0, o1;
#pragma unroll
      for (int r = 0; r < 4; ++r) o0[r] = acc[r] + c0[r];
#pragma unroll
      for (int r = 0; r < 4; ++r) o1[r] = acc[r + 4] + c1[r];
      float* dst = outp + (rm + l15) * kC + n0 + (hi << 3);
      *(v4f*)dst = o0;
      *(v4f*)(dst + 4) = o1;
    };

    for (int nt = 0; nt < 8; nt += 2) {
      v8f a0 = y_tile(nt);
      v8f a1 = y_tile(nt + 1);
      y_epi(nt, a0);
      y_epi(nt + 1, a1);
    }
  }
}

extern "C" void kernel_launch(void* const* d_in, const int* in_sizes, int n_in,
                              void* d_out, int out_size, void* d_ws, size_t ws_size,
                              hipStream_t stream) {
  const float* x    = (const float*)d_in[0];
  const float* Wqkv = (const float*)d_in[1];
  const float* bqkv = (const float*)d_in[2];
  const float* Wp   = (const float*)d_in[3];
  const float* bp   = (const float*)d_in[4];
  float* out = (float*)d_out;

  const int chunks = in_sizes[0] / (kChunk * kC);  // 4096 for B=16, N=16384
  const int blocks = chunks / 2;                   // 2 chunks per workgroup

  (void)hipFuncSetAttribute(reinterpret_cast<const void*>(block_attn_kernel),
                            hipFuncAttributeMaxDynamicSharedMemorySize,
                            (int)kSmemBytes);
  block_attn_kernel<<<blocks, 256, kSmemBytes, stream>>>(x, Wqkv, bqkv, Wp, bp, out);
}
